// MultiHeadAttention_26688926778087
// MI455X (gfx1250) — matrix operand order repaired and checked by reading, not compiled, not run on hardware
//
#include <hip/hip_runtime.h>

#define DMODEL 1024
#define SEQ    2048
#define NHEAD  16
#define DHEAD  64

typedef __attribute__((ext_vector_type(16))) __bf16 v16bf;
typedef __attribute__((ext_vector_type(8)))  float  v8f;
typedef unsigned int v4u __attribute__((ext_vector_type(4)));
typedef int          v8i __attribute__((ext_vector_type(8)));
typedef int          v4i __attribute__((ext_vector_type(4)));

union FragBF {
  v16bf v;
  uint4 u[2];
};

__device__ __forceinline__ __bf16 f2bf(float x) { return (__bf16)x; }  // v_cvt_pk_bf16_f32
__device__ __forceinline__ unsigned short bfbits(float x) {
  union { __bf16 b; unsigned short u; } c;
  c.b = (__bf16)x;
  return c.u;
}

// LDS byte offset of a __shared__ object (addrspace(3) ptrtoint)
typedef __attribute__((address_space(3))) const void lds_cv;
__device__ __forceinline__ unsigned ldsOffset(const void* p) {
  return (unsigned)(uintptr_t)(lds_cv*)p;
}

// ---------------------------------------------------------------------------
// TDM: DMA one 64x64 bf16 tile (row pitch 64 elems in memory) into LDS with
// +16B pad per 128B row (-> 72-halfword LDS pitch). D# per CDNA5 ISA.
// 6-arg builtin form (clang-23 / therock-10.0 lane).
// ---------------------------------------------------------------------------
__device__ __forceinline__ void tdm_load_tile64x64_bf16(unsigned ldsByteAddr,
                                                        const unsigned short* gsrc) {
  const unsigned long long ga = (unsigned long long)(uintptr_t)gsrc;
  v4u g0;
  g0.x = 1u;                                             // count=1, user mode
  g0.y = ldsByteAddr;                                    // lds_addr
  g0.z = (unsigned)(ga & 0xffffffffu);                   // global_addr[31:0]
  g0.w = (unsigned)((ga >> 32) & 0x01ffffffu) | (2u << 30);  // ga[56:32], type=2
  v8i g1;
  // data_size=2B, pad_enable, pad_interval: 32 dwords (code 4), pad_amount: 4 dwords (code 3)
  g1[0] = (1 << 16) | (1 << 20) | (4 << 22) | (3 << 25);
  g1[1] = (64 << 16);                                    // tensor_dim0 = 64
  g1[2] = (int)(2048u << 16);                            // tensor_dim1 = 2048
  g1[3] = (64 << 16);                                    // tile_dim0 = 64
  g1[4] = 64;                                            // tile_dim1 = 64
  g1[5] = 64;                                            // tensor_dim0_stride = 64
  g1[6] = 0;
  g1[7] = 0;
  const v4i z4 = (v4i){0, 0, 0, 0};                      // groups 2/3 unused (2D)
  const v8i z8 = (v8i){0, 0, 0, 0, 0, 0, 0, 0};
  __builtin_amdgcn_tensor_load_to_lds(g0, g1, z4, z4, z8, 0);
}

// ---------------------------------------------------------------------------
// GEMM: Y = X @ W^T + bias   (X fp32 [M,1024] rm, W fp32 [1024,1024] rm)
// 128 thr / 4 waves, 64x64 tile, K-step 32, double-buffered LDS (1 barrier),
// register prefetch of next tile + global_prefetch of t+2.
// mode 0: fp32 [M,1024] row-major   (output projection -> d_out)
// mode 1: bf16 head-split [B,H,S,64] (K/V projections)
// mode 2: bf16 head-split, value pre-scaled by 1/sqrt(64) (Q projection)
// ---------------------------------------------------------------------------
#define GSTR 40  // LDS row stride in bf16 (32 + 8 pad)

__global__ __launch_bounds__(128)
void gemm_bias_kernel(const float* __restrict__ X,
                      const float* __restrict__ W,
                      const float* __restrict__ bias,
                      void* __restrict__ Y,
                      int mode) {
  __shared__ alignas(16) __bf16 Xs[2][64 * GSTR];
  __shared__ alignas(16) __bf16 Ws[2][64 * GSTR];

  const int tid   = threadIdx.x;
  const int wv    = tid >> 5;
  const int lane  = tid & 31;
  const int ln16  = lane & 15;
  const int grp   = lane >> 4;
  const int mbase = blockIdx.x * 64;
  const int nbase = blockIdx.y * 64;

  int srow[4], scol[4];
#pragma unroll
  for (int i = 0; i < 4; ++i) {
    const int e4 = tid + i * 128;       // 512 float4 per 64x32 tile
    srow[i] = e4 >> 3;
    scol[i] = (e4 & 7) << 2;
  }

  v8f acc[4];
#pragma unroll
  for (int j = 0; j < 4; ++j) acc[j] = (v8f){0.f,0.f,0.f,0.f,0.f,0.f,0.f,0.f};

  float4 xr[4], wr[4];
#pragma unroll
  for (int i = 0; i < 4; ++i) {
    xr[i] = *(const float4*)(X + (size_t)(mbase + srow[i]) * DMODEL + scol[i]);
    wr[i] = *(const float4*)(W + (size_t)(nbase + srow[i]) * DMODEL + scol[i]);
  }
#pragma unroll
  for (int i = 0; i < 4; ++i) {
    __bf16* xd = &Xs[0][srow[i] * GSTR + scol[i]];
    xd[0] = f2bf(xr[i].x); xd[1] = f2bf(xr[i].y);
    xd[2] = f2bf(xr[i].z); xd[3] = f2bf(xr[i].w);
    __bf16* wd = &Ws[0][srow[i] * GSTR + scol[i]];
    wd[0] = f2bf(wr[i].x); wd[1] = f2bf(wr[i].y);
    wd[2] = f2bf(wr[i].z); wd[3] = f2bf(wr[i].w);
  }

  const int arow = wv * 16 + ln16;
  for (int t = 0; t < DMODEL / 32; ++t) {
    const int cur = t & 1, nxt = cur ^ 1;
    __syncthreads();

    const int kn = (t + 1) * 32;
    if (kn < DMODEL) {
#pragma unroll
      for (int i = 0; i < 4; ++i) {
        xr[i] = *(const float4*)(X + (size_t)(mbase + srow[i]) * DMODEL + kn + scol[i]);
        wr[i] = *(const float4*)(W + (size_t)(nbase + srow[i]) * DMODEL + kn + scol[i]);
      }
      if (kn + 32 < DMODEL) {
        __builtin_prefetch(X + (size_t)(mbase + srow[0]) * DMODEL + kn + 32 + scol[0], 0, 1);
        __builtin_prefetch(W + (size_t)(nbase + srow[0]) * DMODEL + kn + 32 + scol[0], 0, 1);
      }
    }

    FragBF a;
    a.u[0] = *(const uint4*)(&Xs[cur][arow * GSTR + 8 * grp]);
    a.u[1] = *(const uint4*)(&Xs[cur][arow * GSTR + 8 * grp + 16]);
#pragma unroll
    for (int j = 0; j < 4; ++j) {
      FragBF bm;
      const int brow = j * 16 + ln16;
      bm.u[0] = *(const uint4*)(&Ws[cur][brow * GSTR + 8 * grp]);
      bm.u[1] = *(const uint4*)(&Ws[cur][brow * GSTR + 8 * grp + 16]);
      acc[j] = __builtin_amdgcn_wmma_f32_16x16x32_bf16(
          false, a.v, false, bm.v, (short)0, acc[j], false, false);
    }

    if (kn < DMODEL) {
#pragma unroll
      for (int i = 0; i < 4; ++i) {
        __bf16* xd = &Xs[nxt][srow[i] * GSTR + scol[i]];
        xd[0] = f2bf(xr[i].x); xd[1] = f2bf(xr[i].y);
        xd[2] = f2bf(xr[i].z); xd[3] = f2bf(xr[i].w);
        __bf16* wd = &Ws[nxt][srow[i] * GSTR + scol[i]];
        wd[0] = f2bf(wr[i].x); wd[1] = f2bf(wr[i].y);
        wd[2] = f2bf(wr[i].z); wd[3] = f2bf(wr[i].w);
      }
    }
  }

#pragma unroll
  for (int j = 0; j < 4; ++j) {
    const int gn = nbase + j * 16 + ln16;
    const float bv = bias[gn];
#pragma unroll
    for (int vv = 0; vv < 8; ++vv) {
      const int gm = mbase + wv * 16 + vv + 8 * grp;
      const float val = acc[j][vv] + bv;
      if (mode == 0) {
        ((float*)Y)[(size_t)gm * DMODEL + gn] = val;
      } else {
        const int b = gm >> 11, s = gm & 2047;
        const int hh = gn >> 6, d = gn & 63;
        const float sv = (mode == 2) ? val * 0.125f : val;
        ((unsigned short*)Y)[((size_t)((b * NHEAD + hh) * SEQ + s)) * DHEAD + d] = bfbits(sv);
      }
    }
  }
}

// ---------------------------------------------------------------------------
// Flash attention over bf16 head-split Q/K/V. One block = (b, h, 64-row Q
// tile); 4 waves x 16 query rows. Q/K tiles staged by the Tensor Data Mover
// (tensor_load_to_lds + s_wait_tensorcnt); V staged transposed by VALU.
// ---------------------------------------------------------------------------
#define ASTR 72  // LDS row stride in bf16 (64 + 8 pad; matches TDM pad)

__global__ __launch_bounds__(128)
void attn_kernel(const unsigned short* __restrict__ Qhb,
                 const unsigned short* __restrict__ Khb,
                 const unsigned short* __restrict__ Vhb,
                 float* __restrict__ Ctx) {
  __shared__ alignas(16) unsigned short Qs[64 * ASTR];
  __shared__ alignas(16) unsigned short Ks[64 * ASTR];
  __shared__ alignas(16) unsigned short Vt[64 * ASTR];  // transposed: [d][key]
  __shared__ alignas(16) unsigned short Ps[64 * ASTR];

  const int tid  = threadIdx.x;
  const int wv   = tid >> 5;
  const int lane = tid & 31;
  const int ln16 = lane & 15;
  const int grp  = lane >> 4;
  const int qt = blockIdx.x, hh = blockIdx.y, bb = blockIdx.z;
  const size_t headOff = ((size_t)(bb * NHEAD + hh)) * SEQ * DHEAD;
  const int qbase = qt * 64;

  // Q tile via TDM (already bf16 + pre-scaled by 1/sqrt(64) at projection)
  if (wv == 0)
    tdm_load_tile64x64_bf16(ldsOffset(Qs), Qhb + headOff + (size_t)qbase * DHEAD);

  v8f o[4];
#pragma unroll
  for (int j = 0; j < 4; ++j) o[j] = (v8f){0.f,0.f,0.f,0.f,0.f,0.f,0.f,0.f};
  float m_i[8], l_i[8], al[8], rs[8];
#pragma unroll
  for (int vv = 0; vv < 8; ++vv) { m_i[vv] = -3.0e38f; l_i[vv] = 0.f; }

  const int arow = wv * 16 + ln16;

  for (int kt = 0; kt <= qt; ++kt) {
    const int kbase = kt * 64;
    __syncthreads();  // previous iteration done reading Ks/Vt

    // K tile via TDM; V tile transposed by VALU (bf16 -> bf16, no converts)
    if (wv == 0)
      tdm_load_tile64x64_bf16(ldsOffset(Ks), Khb + headOff + (size_t)kbase * DHEAD);
#pragma unroll
    for (int i = 0; i < 4; ++i) {
      const int e8 = tid + i * 128;     // 512 x (8 bf16) per 64x64 tile
      const int key = e8 >> 3, d8 = (e8 & 7) << 3;
      uint4 vv4 = *(const uint4*)(Vhb + headOff + (size_t)(kbase + key) * DHEAD + d8);
      const unsigned short* pv = (const unsigned short*)&vv4;
#pragma unroll
      for (int u = 0; u < 8; ++u) Vt[(d8 + u) * ASTR + key] = pv[u];
    }
    if (wv == 0) __builtin_amdgcn_s_wait_tensorcnt(0);
    __syncthreads();  // Qs/Ks (TDM) and Vt visible to all waves

    // scores: S = Qs(16x64 strip) @ Ks^T, contraction over d
    v8f s[4];
#pragma unroll
    for (int j = 0; j < 4; ++j) s[j] = (v8f){0.f,0.f,0.f,0.f,0.f,0.f,0.f,0.f};
#pragma unroll
    for (int k0 = 0; k0 < DHEAD; k0 += 32) {
      FragBF a;
      a.u[0] = *(const uint4*)(&Qs[arow * ASTR + k0 + 8 * grp]);
      a.u[1] = *(const uint4*)(&Qs[arow * ASTR + k0 + 8 * grp + 16]);
#pragma unroll
      for (int j = 0; j < 4; ++j) {
        FragBF bm;
        const int brow = j * 16 + ln16;
        bm.u[0] = *(const uint4*)(&Ks[brow * ASTR + k0 + 8 * grp]);
        bm.u[1] = *(const uint4*)(&Ks[brow * ASTR + k0 + 8 * grp + 16]);
        s[j] = __builtin_amdgcn_wmma_f32_16x16x32_bf16(
            false, a.v, false, bm.v, (short)0, s[j], false, false);
      }
    }

    // causal mask on diagonal tile
    if (kt == qt) {
#pragma unroll
      for (int j = 0; j < 4; ++j)
#pragma unroll
        for (int vv = 0; vv < 8; ++vv) {
          const int qloc = wv * 16 + vv + 8 * grp;
          const int kloc = j * 16 + ln16;
          if (kloc > qloc) s[j][vv] = -1.0e9f;
        }
    }

    // online softmax (16-lane shuffle reductions; C/D row = v + 8*grp)
    float mx[8];
#pragma unroll
    for (int vv = 0; vv < 8; ++vv)
      mx[vv] = fmaxf(fmaxf(s[0][vv], s[1][vv]), fmaxf(s[2][vv], s[3][vv]));
#pragma unroll
    for (int off = 1; off <= 8; off <<= 1)
#pragma unroll
      for (int vv = 0; vv < 8; ++vv)
        mx[vv] = fmaxf(mx[vv], __shfl_xor(mx[vv], off));
#pragma unroll
    for (int vv = 0; vv < 8; ++vv) {
      const float mn = fmaxf(m_i[vv], mx[vv]);
      al[vv] = __expf(m_i[vv] - mn);
      m_i[vv] = mn;
      rs[vv] = 0.f;
    }
#pragma unroll
    for (int j = 0; j < 4; ++j)
#pragma unroll
      for (int vv = 0; vv < 8; ++vv) {
        const float p = __expf(s[j][vv] - m_i[vv]);
        rs[vv] += p;
        Ps[(wv * 16 + vv + 8 * grp) * ASTR + j * 16 + ln16] = bfbits(p);
      }
#pragma unroll
    for (int off = 1; off <= 8; off <<= 1)
#pragma unroll
      for (int vv = 0; vv < 8; ++vv)
        rs[vv] += __shfl_xor(rs[vv], off);
#pragma unroll
    for (int vv = 0; vv < 8; ++vv) l_i[vv] = l_i[vv] * al[vv] + rs[vv];
#pragma unroll
    for (int j = 0; j < 4; ++j)
#pragma unroll
      for (int vv = 0; vv < 8; ++vv) o[j][vv] *= al[vv];

    __syncthreads();  // Ps visible across lanes

    // O += P(16x64 strip) @ V(64x64), contraction over key dim
#pragma unroll
    for (int k0 = 0; k0 < 64; k0 += 32) {
      FragBF a;
      a.u[0] = *(const uint4*)(&Ps[arow * ASTR + k0 + 8 * grp]);
      a.u[1] = *(const uint4*)(&Ps[arow * ASTR + k0 + 8 * grp + 16]);
#pragma unroll
      for (int j = 0; j < 4; ++j) {
        FragBF bm;  // B(k=key, n=d) = Vt[d][key]
        const int brow = j * 16 + ln16;
        bm.u[0] = *(const uint4*)(&Vt[brow * ASTR + k0 + 8 * grp]);
        bm.u[1] = *(const uint4*)(&Vt[brow * ASTR + k0 + 8 * grp + 16]);
        o[j] = __builtin_amdgcn_wmma_f32_16x16x32_bf16(
            false, a.v, false, bm.v, (short)0, o[j], false, false);
      }
    }
  }

  // normalize and write merged-head ctx: [B, S, H*64] fp32 row-major
#pragma unroll
  for (int vv = 0; vv < 8; ++vv) {
    const float inv = 1.0f / l_i[vv];
    const int sloc = qbase + wv * 16 + vv + 8 * grp;
#pragma unroll
    for (int j = 0; j < 4; ++j) {
      const int d = j * 16 + ln16;
      Ctx[((size_t)(bb * SEQ + sloc)) * DMODEL + hh * DHEAD + d] = o[j][vv] * inv;
    }
  }
}

// ---------------------------------------------------------------------------
extern "C" void kernel_launch(void* const* d_in, const int* in_sizes, int n_in,
                              void* d_out, int out_size, void* d_ws, size_t ws_size,
                              hipStream_t stream) {
  (void)in_sizes; (void)n_in; (void)out_size; (void)ws_size;
  const float* q  = (const float*)d_in[0];
  const float* k  = (const float*)d_in[1];
  const float* v  = (const float*)d_in[2];
  // d_in[3] = mask (causal; computed analytically in-kernel)
  const float* Wq = (const float*)d_in[4];
  const float* bq = (const float*)d_in[5];
  const float* Wk = (const float*)d_in[6];
  const float* bk = (const float*)d_in[7];
  const float* Wv = (const float*)d_in[8];
  const float* bv = (const float*)d_in[9];
  const float* Wo = (const float*)d_in[10];
  const float* bo = (const float*)d_in[11];
  float* out = (float*)d_out;

  const size_t TS = (size_t)2 * SEQ * DMODEL;          // elements per tensor
  unsigned short* Qh = (unsigned short*)d_ws;          // bf16 [B,H,S,64], prescaled
  unsigned short* Kh = Qh + TS;                        // bf16 [B,H,S,64]
  unsigned short* Vh = Kh + TS;                        // bf16 [B,H,S,64]
  float*          Cx = (float*)(Vh + TS);              // fp32 [B,S,1024]

  dim3 gg(64, 16), blk(128);
  gemm_bias_kernel<<<gg, blk, 0, stream>>>(q, Wq, bq, Qh, 2);
  gemm_bias_kernel<<<gg, blk, 0, stream>>>(k, Wk, bk, Kh, 1);
  gemm_bias_kernel<<<gg, blk, 0, stream>>>(v, Wv, bv, Vh, 1);
  attn_kernel<<<dim3(SEQ / 64, NHEAD, 2), blk, 0, stream>>>(Qh, Kh, Vh, Cx);
  gemm_bias_kernel<<<gg, blk, 0, stream>>>(Cx, Wo, bo, out, 0);
}
